// SFIMLLIE_22342419874204
// MI455X (gfx1250) — matrix operand order repaired and checked by reading, not compiled, not run on hardware
//
#include <hip/hip_runtime.h>
#include <hip/hip_bf16.h>

// ---------------- problem constants ----------------
#define B_  2
#define H_  64
#define W_  64
#define C_  96
#define D_  192
#define N_  16
#define R_  6
#define K_  4
#define E_  3
#define L_  (H_ * W_)          // 4096
#define CP_ (R_ + 2 * N_)      // 38

typedef __attribute__((ext_vector_type(16))) _Float16 v16h;
typedef __attribute__((ext_vector_type(8)))  float    v8f;

static inline int cdiv(int a, int b) { return (a + b - 1) / b; }

// ---------------- activation helpers ----------------
__device__ __forceinline__ float act_apply(float v, int act) {
  switch (act) {
    case 1: return v > 0.f ? v : 0.2f * v;                         // leaky 0.2
    case 2: return 1.f / (1.f + __expf(-v));                       // sigmoid
    case 3: return (v > 20.f) ? v : log1pf(__expf(v));             // softplus
    case 4: return 0.5f * v * (1.f + erff(v * 0.70710678118f));    // gelu (exact)
  }
  return v;
}

// ======================================================================
// Batched GEMM, one wave -> 16x32 output strip (2 accumulators, shared A
// fragment), v_wmma_f32_16x16x32_f16.
//   C[z][m,n] = act( (sum_k A[z][m,k]*B[zb][n,k] + bias[zb][n]) * rowScale )
// Template APLANAR fixes the A K-stride at compile time (1 or L_), so
// fragment loads are immediate-offset clauses off a single base address.
// B is always row-major (N,K): 4x b128 loads per fragment.
// OOB M/N lanes use clamped addresses (stores are predicated); only the
// K remainder is value-masked (cndmask, no EXEC branching).
// ======================================================================
template <bool APLANAR>
__global__ void wmma_gemm_t(
    const float* __restrict__ A, long sAm, long sAb,
    const float* __restrict__ Bw, long sBn, long sBb, int bMod,
    float* __restrict__ C, long sCm, long sCn, long sCb,
    const float* __restrict__ bias, long sBiasB, int biasMod,
    const float* __restrict__ rowScale, long sRsB,
    int M, int N, int K, int act)
{
  const long sAk = APLANAR ? (long)L_ : 1L;
  const int lane = threadIdx.x & 31;
  const int wid  = threadIdx.x >> 5;
  const int tilesN  = (N + 15) >> 4;
  const int tilesM  = (M + 15) >> 4;
  const int tilesNP = (tilesN + 1) >> 1;            // N-tile pairs
  const int tile = blockIdx.x * (blockDim.x >> 5) + wid;
  if (tile >= tilesM * tilesNP) return;
  const int tm  = tile / tilesNP;
  const int tn0 = (tile % tilesNP) * 2;
  const int tn1 = tn0 + 1;
  const int z = blockIdx.z;

  const float* Ab = A + (long)z * sAb;
  const float* Bb = Bw + (long)(bMod > 0 ? (z % bMod) : z) * sBb;
  float*       Cb = C + (long)z * sCb;

  const int hi = lane >> 4;                          // lane half-group
  const int mrow0  = tm * 16 + (lane & 15);
  const int mrow   = mrow0 < M ? mrow0 : M - 1;      // address clamp only
  const int ncolA0 = tn0 * 16 + (lane & 15);
  const int ncolA1 = tn1 * 16 + (lane & 15);
  const int ncol0  = ncolA0 < N ? ncolA0 : N - 1;
  const int ncol1  = ncolA1 < N ? ncolA1 : N - 1;

  const float* aBase  = Ab + (long)mrow * sAm + (long)hi * 16 * sAk;
  const float* bBase0 = Bb + (long)ncol0 * sBn + (long)hi * 16;
  const float* bBase1 = Bb + (long)ncol1 * sBn + (long)hi * 16;

  v8f acc0 = {}, acc1 = {};
  for (int kk = 0; kk < K; kk += 32) {
    v16h av, bv0, bv1;
    if (kk + 32 <= K) {
      // ---------------- full K step: unconditional loads ----------------
      if (APLANAR) {
        const float* p = aBase + (long)kk * (long)L_;
#pragma unroll
        for (int i = 0; i < 16; ++i)     // compile-time offsets -> IOFFSET
          av[i] = (_Float16)p[(long)i * (long)L_];
      } else {
        const float4* p = (const float4*)(aBase + kk);
        float4 q0 = p[0], q1 = p[1], q2 = p[2], q3 = p[3]; // k..k+15
        av[0]=(_Float16)q0.x; av[1]=(_Float16)q0.y; av[2]=(_Float16)q0.z; av[3]=(_Float16)q0.w;
        av[4]=(_Float16)q1.x; av[5]=(_Float16)q1.y; av[6]=(_Float16)q1.z; av[7]=(_Float16)q1.w;
        av[8]=(_Float16)q2.x; av[9]=(_Float16)q2.y; av[10]=(_Float16)q2.z; av[11]=(_Float16)q2.w;
        av[12]=(_Float16)q3.x; av[13]=(_Float16)q3.y; av[14]=(_Float16)q3.z; av[15]=(_Float16)q3.w;
      }
      {
        const float4* p = (const float4*)(bBase0 + kk);   // 16 consecutive k
        float4 q0 = p[0], q1 = p[1], q2 = p[2], q3 = p[3];
        bv0[0]=(_Float16)q0.x; bv0[1]=(_Float16)q0.y; bv0[2]=(_Float16)q0.z; bv0[3]=(_Float16)q0.w;
        bv0[4]=(_Float16)q1.x; bv0[5]=(_Float16)q1.y; bv0[6]=(_Float16)q1.z; bv0[7]=(_Float16)q1.w;
        bv0[8]=(_Float16)q2.x; bv0[9]=(_Float16)q2.y; bv0[10]=(_Float16)q2.z; bv0[11]=(_Float16)q2.w;
        bv0[12]=(_Float16)q3.x; bv0[13]=(_Float16)q3.y; bv0[14]=(_Float16)q3.z; bv0[15]=(_Float16)q3.w;
      }
      {
        const float4* p = (const float4*)(bBase1 + kk);
        float4 q0 = p[0], q1 = p[1], q2 = p[2], q3 = p[3];
        bv1[0]=(_Float16)q0.x; bv1[1]=(_Float16)q0.y; bv1[2]=(_Float16)q0.z; bv1[3]=(_Float16)q0.w;
        bv1[4]=(_Float16)q1.x; bv1[5]=(_Float16)q1.y; bv1[6]=(_Float16)q1.z; bv1[7]=(_Float16)q1.w;
        bv1[8]=(_Float16)q2.x; bv1[9]=(_Float16)q2.y; bv1[10]=(_Float16)q2.z; bv1[11]=(_Float16)q2.w;
        bv1[12]=(_Float16)q3.x; bv1[13]=(_Float16)q3.y; bv1[14]=(_Float16)q3.z; bv1[15]=(_Float16)q3.w;
      }
      if (kk + 64 <= K) {
        __builtin_prefetch(aBase + (long)(kk + 32) * sAk, 0, 1);
        __builtin_prefetch(bBase0 + (kk + 32), 0, 1);
      }
    } else {
      // -------- K remainder: clamped addresses + value mask (no EXEC) ----
      float ta[16], tb0[16], tb1[16];
#pragma unroll
      for (int i = 0; i < 16; ++i) {
        int k  = kk + i + (hi << 4);
        int kc = k < K ? k : K - 1;
        float va = aBase[((long)kc - hi * 16) * sAk];
        ta[i] = (k < K) ? va : 0.f;
      }
#pragma unroll
      for (int i = 0; i < 16; ++i) {
        int k  = kk + i + (hi << 4);
        int kc = k < K ? k : K - 1;
        float v0 = bBase0[(long)kc - hi * 16];
        float v1 = bBase1[(long)kc - hi * 16];
        tb0[i] = (k < K) ? v0 : 0.f;
        tb1[i] = (k < K) ? v1 : 0.f;
      }
#pragma unroll
      for (int i = 0; i < 16; ++i) {
        av[i] = (_Float16)ta[i]; bv0[i] = (_Float16)tb0[i]; bv1[i] = (_Float16)tb1[i];
      }
    }
    acc0 = __builtin_amdgcn_wmma_f32_16x16x32_f16(
        false, av, false, bv0, (short)0, acc0, false, false);
    acc1 = __builtin_amdgcn_wmma_f32_16x16x32_f16(
        false, av, false, bv1, (short)0, acc1, false, false);
  }

  const float* biasB =
      bias ? bias + (long)(biasMod > 0 ? (z % biasMod) : z) * sBiasB : nullptr;
  const float biasV0 = biasB ? biasB[ncol0] : 0.f;
  const float biasV1 = biasB ? biasB[ncol1] : 0.f;
#pragma unroll
  for (int r = 0; r < 8; ++r) {
    int m = tm * 16 + r + (hi << 3);   // C layout: VGPR r -> rows r, r+8
    if (m < M) {
      float rsv = rowScale ? rowScale[(long)z * sRsB + m] : 1.f;
      if (ncolA0 < N)
        Cb[(long)m * sCm + (long)ncolA0 * sCn] =
            act_apply((acc0[r] + biasV0) * rsv, act);
      if (ncolA1 < N)
        Cb[(long)m * sCm + (long)ncolA1 * sCn] =
            act_apply((acc1[r] + biasV1) * rsv, act);
    }
  }
}

// ======================================================================
// Depthwise 3x3 conv ('SAME'), generic input strides, planar (B,D,L) output.
// act: 0 none, 1 silu, 2 gelu
// ======================================================================
__global__ void dwconv3x3_kernel(const float* __restrict__ in, long inB, long inC, long inP,
                                 const float* __restrict__ w, const float* __restrict__ bias,
                                 float* __restrict__ out, int act)
{
  int idx = blockIdx.x * blockDim.x + threadIdx.x;
  if (idx >= B_ * D_ * L_) return;
  int p = idx % L_;
  int c = (idx / L_) % D_;
  int b = idx / (D_ * L_);
  int h = p / W_, x = p % W_;
  const float* wb = w + c * 9;
  float acc = bias ? bias[c] : 0.f;
#pragma unroll
  for (int dy = -1; dy <= 1; ++dy)
#pragma unroll
    for (int dx = -1; dx <= 1; ++dx) {
      int hh = h + dy, ww = x + dx;
      if (hh >= 0 && hh < H_ && ww >= 0 && ww < W_)
        acc += wb[(dy + 1) * 3 + (dx + 1)] *
               in[(long)b * inB + (long)c * inC + (long)(hh * W_ + ww) * inP];
    }
  float v = acc;
  if (act == 1)      v = acc / (1.f + __expf(-acc));
  else if (act == 2) v = 0.5f * acc * (1.f + erff(acc * 0.70710678118f));
  out[idx] = v;
}

// 3x3 conv from a single input channel -> Co channels (ig1 / router1)
__global__ void conv3x3_c1_kernel(const float* __restrict__ in /*(B,L)*/,
                                  const float* __restrict__ w /*(Co,1,3,3)*/,
                                  const float* __restrict__ bias,
                                  float* __restrict__ out /*(B,Co,L)*/,
                                  int Co, int act)
{
  int idx = blockIdx.x * blockDim.x + threadIdx.x;
  if (idx >= B_ * Co * L_) return;
  int p = idx % L_;
  int c = (idx / L_) % Co;
  int b = idx / (Co * L_);
  int h = p / W_, x = p % W_;
  const float* wb = w + c * 9;
  float acc = bias[c];
#pragma unroll
  for (int dy = -1; dy <= 1; ++dy)
#pragma unroll
    for (int dx = -1; dx <= 1; ++dx) {
      int hh = h + dy, ww = x + dx;
      if (hh >= 0 && hh < H_ && ww >= 0 && ww < W_)
        acc += wb[(dy + 1) * 3 + (dx + 1)] * in[(long)b * L_ + hh * W_ + ww];
    }
  out[idx] = act_apply(acc, act);
}

// I1 = mean over 3 channels of illum (B,3,H,W)
__global__ void illum_mean_kernel(const float* __restrict__ illum, float* __restrict__ I1)
{
  int idx = blockIdx.x * blockDim.x + threadIdx.x;
  if (idx >= B_ * L_) return;
  int p = idx % L_, b = idx / L_;
  const float* base = illum + (long)b * 3 * L_ + p;
  I1[idx] = (base[0] + base[L_] + base[2 * L_]) * (1.f / 3.f);
}

// in-place softmax over the channel dim of a planar (B,Cn,L) tensor
__global__ void softmax_ch_kernel(float* __restrict__ data, int Cn)
{
  int idx = blockIdx.x * blockDim.x + threadIdx.x;
  if (idx >= B_ * L_) return;
  int p = idx % L_, b = idx / L_;
  float* base = data + (long)b * Cn * L_ + p;
  float mx = -3.4e38f;
  for (int c = 0; c < Cn; ++c) mx = fmaxf(mx, base[(long)c * L_]);
  float s = 0.f;
  for (int c = 0; c < Cn; ++c) s += __expf(base[(long)c * L_] - mx);
  float inv = 1.f / s;
  for (int c = 0; c < Cn; ++c)
    base[(long)c * L_] = __expf(base[(long)c * L_] - mx) * inv;
}

__global__ void mul_kernel(const float* __restrict__ a, const float* __restrict__ b,
                           float* __restrict__ o, int n)
{
  int idx = blockIdx.x * blockDim.x + threadIdx.x;
  if (idx < n) o[idx] = a[idx] * b[idx];
}

// cross-scan gather: xs[b,k,d,l] = xconv[b,d,scan[k,l]] ; struct[b,k,l] = sm[b,k,scan[k,l]]
__global__ void cross_scan_kernel(const float* __restrict__ xconv, const float* __restrict__ sm,
                                  const int* __restrict__ scan_ids,
                                  float* __restrict__ xs, float* __restrict__ structS)
{
  long idx = (long)blockIdx.x * blockDim.x + threadIdx.x;
  if (idx >= (long)B_ * K_ * D_ * L_) return;
  int l = idx % L_;
  int d = (idx / L_) % D_;
  int k = (idx / ((long)D_ * L_)) % K_;
  int b = idx / ((long)K_ * D_ * L_);
  int src = scan_ids[k * L_ + l];
  xs[idx] = xconv[((long)b * D_ + d) * L_ + src];
  if (d == 0)
    structS[((long)b * K_ + k) * L_ + l] = sm[((long)b * K_ + k) * L_ + src];
}

// ======================================================================
// Selective scan: one block per (b,k), 192 threads = d-channels.
// h-state (16 f32) in registers; Bs/Cs chunked through LDS.
// ======================================================================
#define SCHUNK 64
__global__ void selective_scan_kernel(const float* __restrict__ xs,
                                      const float* __restrict__ delta,
                                      const float* __restrict__ xdbl,
                                      const float* __restrict__ A_logs,
                                      const float* __restrict__ Ds,
                                      float* __restrict__ ys)
{
  const int z = blockIdx.x;        // b*K + k
  const int k = z % K_;
  const int d = threadIdx.x;       // 0..191
  __shared__ float sB[SCHUNK][N_];
  __shared__ float sC[SCHUNK][N_];

  float Areg[N_], h[N_];
  float Dd = Ds[k * D_ + d];
#pragma unroll
  for (int n = 0; n < N_; ++n) {
    Areg[n] = -__expf(A_logs[((long)k * D_ + d) * N_ + n]);
    h[n] = 0.f;
  }
  const float* Bsrc = xdbl + (long)z * CP_ * L_ + (long)R_ * L_;
  const float* Csrc = Bsrc + (long)N_ * L_;
  const float* dl = delta + ((long)z * D_ + d) * L_;
  const float* xl = xs    + ((long)z * D_ + d) * L_;
  float*       yl = ys    + ((long)z * D_ + d) * L_;

  for (int l0 = 0; l0 < L_; l0 += SCHUNK) {
    for (int idx = threadIdx.x; idx < SCHUNK * N_; idx += blockDim.x) {
      int n = idx / SCHUNK, t = idx % SCHUNK;      // coalesced over t
      sB[t][n] = Bsrc[(long)n * L_ + l0 + t];
      sC[t][n] = Csrc[(long)n * L_ + l0 + t];
    }
    __syncthreads();
    for (int t = 0; t < SCHUNK; ++t) {
      float dt = dl[l0 + t];
      float xv = xl[l0 + t];
      float dx = dt * xv;
      float yv = 0.f;
#pragma unroll
      for (int n = 0; n < N_; ++n) {
        h[n] = h[n] * __expf(dt * Areg[n]) + dx * sB[t][n];
        yv += h[n] * sC[t][n];
      }
      yl[l0 + t] = yv + Dd * xv;
    }
    __syncthreads();
  }
}

// inverse gather + sum over K: ysum[b,l,d] = sum_k ys[b,k,d,inv[k,l]]
__global__ void inv_merge_kernel(const float* __restrict__ ys, const int* __restrict__ inv_ids,
                                 float* __restrict__ ysum)
{
  long idx = (long)blockIdx.x * blockDim.x + threadIdx.x;
  if (idx >= (long)B_ * L_ * D_) return;
  int d = idx % D_;
  int l = (idx / D_) % L_;
  int b = idx / ((long)L_ * D_);
  float s = 0.f;
#pragma unroll
  for (int k = 0; k < K_; ++k)
    s += ys[(((long)b * K_ + k) * D_ + d) * L_ + inv_ids[k * L_ + l]];
  ysum[idx] = s;
}

// LayerNorm over D + silu(z) gate; one block per (b,l) row
__global__ void ln_gate_kernel(const float* __restrict__ ysum,
                               const float* __restrict__ g, const float* __restrict__ bb,
                               const float* __restrict__ xz, float* __restrict__ yact)
{
  const int row = blockIdx.x;          // 0..B*L-1
  const int tid = threadIdx.x;         // 256
  __shared__ float red[256];
  float v = (tid < D_) ? ysum[(long)row * D_ + tid] : 0.f;
  red[tid] = v;
  __syncthreads();
  for (int s = 128; s > 0; s >>= 1) { if (tid < s) red[tid] += red[tid + s]; __syncthreads(); }
  float mean = red[0] * (1.f / D_);
  __syncthreads();
  float c = (tid < D_) ? v - mean : 0.f;
  red[tid] = c * c;
  __syncthreads();
  for (int s = 128; s > 0; s >>= 1) { if (tid < s) red[tid] += red[tid + s]; __syncthreads(); }
  float var = red[0] * (1.f / D_);
  if (tid < D_) {
    float zn = xz[(long)row * (2 * D_) + D_ + tid];
    float sz = zn / (1.f + __expf(-zn));
    yact[(long)row * D_ + tid] = (c * rsqrtf(var + 1e-5f) * g[tid] + bb[tid]) * sz;
  }
}

// fused 3-expert depthwise 3x3 MoE: Fout = Ff + (sum_e alpha_e * dwconv_e(Ff)) * Ff
__global__ void moe_expert_kernel(const float* __restrict__ Ff, const float* __restrict__ alpha,
                                  const float* __restrict__ ew, const float* __restrict__ eb,
                                  float* __restrict__ Fout)
{
  int idx = blockIdx.x * blockDim.x + threadIdx.x;
  if (idx >= B_ * D_ * L_) return;
  int p = idx % L_;
  int c = (idx / L_) % D_;
  int b = idx / (D_ * L_);
  int h = p / W_, x = p % W_;
  float nb[9];
#pragma unroll
  for (int j = 0; j < 9; ++j) {
    int dy = j / 3 - 1, dx = j % 3 - 1;
    int hh = h + dy, ww = x + dx;
    nb[j] = (hh >= 0 && hh < H_ && ww >= 0 && ww < W_)
                ? Ff[((long)b * D_ + c) * L_ + hh * W_ + ww] : 0.f;
  }
  float fm = 0.f;
#pragma unroll
  for (int e = 0; e < E_; ++e) {
    float acc = eb[e * D_ + c];
    const float* wj = ew + ((long)e * D_ + c) * 9;
#pragma unroll
    for (int j = 0; j < 9; ++j) acc += wj[j] * nb[j];
    fm += alpha[((long)b * E_ + e) * L_ + p] * acc;
  }
  float fc = nb[4];
  Fout[idx] = fc + fm * fc;
}

// ======================================================================
// Host orchestration
// ======================================================================
extern "C" void kernel_launch(void* const* d_in, const int* in_sizes, int n_in,
                              void* d_out, int out_size, void* d_ws, size_t ws_size,
                              hipStream_t stream)
{
  (void)in_sizes; (void)n_in; (void)out_size; (void)ws_size;

  const float* x         = (const float*)d_in[0];
  const float* illum     = (const float*)d_in[1];
  const float* in_proj_w = (const float*)d_in[2];
  const float* conv_w    = (const float*)d_in[3];
  const float* conv_b    = (const float*)d_in[4];
  const float* pe1_w     = (const float*)d_in[5];
  const float* pe1_b     = (const float*)d_in[6];
  const float* pe2_w     = (const float*)d_in[7];
  const float* pe2_b     = (const float*)d_in[8];
  const float* ig1_w     = (const float*)d_in[9];
  const float* ig1_b     = (const float*)d_in[10];
  const float* ig2_w     = (const float*)d_in[11];
  const float* ig2_b     = (const float*)d_in[12];
  const float* xprojw    = (const float*)d_in[13];
  const float* dtw       = (const float*)d_in[14];
  const float* dtb       = (const float*)d_in[15];
  const float* A_logs    = (const float*)d_in[16];
  const float* Dsp       = (const float*)d_in[17];
  const float* on_g      = (const float*)d_in[18];
  const float* on_b      = (const float*)d_in[19];
  const float* outw      = (const float*)d_in[20];
  const float* phi_in_w  = (const float*)d_in[21];
  const float* phi_in_b  = (const float*)d_in[22];
  const float* ew        = (const float*)d_in[23];
  const float* eb        = (const float*)d_in[24];
  const float* r1w       = (const float*)d_in[25];
  const float* r1b       = (const float*)d_in[26];
  const float* r2w       = (const float*)d_in[27];
  const float* r2b       = (const float*)d_in[28];
  const float* moeDw     = (const float*)d_in[29];
  const float* moeDb     = (const float*)d_in[30];
  const float* phow      = (const float*)d_in[31];
  const float* phob      = (const float*)d_in[32];
  const int*   scan_ids  = (const int*)d_in[33];
  const int*   inv_ids   = (const int*)d_in[34];
  float* out = (float*)d_out;

  float* ws = (float*)d_ws;
  size_t off = 0;
  auto alloc = [&](size_t n) { float* p = ws + off; off += n; return p; };
  float* xz      = alloc((size_t)B_ * L_ * 2 * D_);
  float* xconv   = alloc((size_t)B_ * D_ * L_);
  float* pemid   = alloc((size_t)B_ * 48 * L_);
  float* phase   = alloc((size_t)B_ * 4 * L_);
  float* igmid   = alloc((size_t)B_ * 48 * L_);
  float* gate    = alloc((size_t)B_ * 4 * L_);
  float* smb     = alloc((size_t)B_ * 4 * L_);
  float* I1      = alloc((size_t)B_ * L_);
  float* xs      = alloc((size_t)B_ * K_ * D_ * L_);
  float* structS = alloc((size_t)B_ * K_ * L_);
  float* xdbl    = alloc((size_t)B_ * K_ * CP_ * L_);
  float* delta   = alloc((size_t)B_ * K_ * D_ * L_);
  float* ysb     = alloc((size_t)B_ * K_ * D_ * L_);
  float* ysum    = alloc((size_t)B_ * L_ * D_);
  float* yact    = alloc((size_t)B_ * L_ * D_);
  float* yout    = alloc((size_t)B_ * L_ * C_);
  float* Ffb     = alloc((size_t)B_ * D_ * L_);
  float* rmid    = alloc((size_t)B_ * 8 * L_);
  float* alphab  = alloc((size_t)B_ * E_ * L_);
  float* Foutb   = alloc((size_t)B_ * D_ * L_);
  float* moemid  = alloc((size_t)B_ * D_ * L_);

  // sAk must be 1 (row-major) or L_ (planar); sBk is always 1.
  auto gemm = [&](const float* A, long sAm, long sAk, long sAb,
                  const float* Bw, long sBn, long sBb, int bMod,
                  float* Cc, long sCm, long sCn, long sCb,
                  const float* bias, long sBiasB, int biasMod,
                  const float* rs, long sRsB,
                  int M, int N, int K, int act, int batch) {
    int tiles = cdiv(M, 16) * cdiv(cdiv(N, 16), 2);
    dim3 grid(cdiv(tiles, 4), 1, batch);
    if (sAk == 1)
      wmma_gemm_t<false><<<grid, 128, 0, stream>>>(
          A, sAm, sAb, Bw, sBn, sBb, bMod,
          Cc, sCm, sCn, sCb, bias, sBiasB, biasMod, rs, sRsB, M, N, K, act);
    else
      wmma_gemm_t<true><<<grid, 128, 0, stream>>>(
          A, sAm, sAb, Bw, sBn, sBb, bMod,
          Cc, sCm, sCn, sCb, bias, sBiasB, biasMod, rs, sRsB, M, N, K, act);
  };
  const float* FNULL = nullptr;

  // 1) in_proj: (B*L,96) x (384,96)^T -> xz (B*L,384)
  gemm(x, C_, 1, 0, in_proj_w, C_, 0, 1,
       xz, 2 * D_, 1, 0, FNULL, 0, 1, FNULL, 0,
       B_ * L_, 2 * D_, C_, 0, 1);

  // 2) depthwise conv3x3 + SiLU on xc (channel-last view of xz) -> xconv (B,D,L)
  dwconv3x3_kernel<<<cdiv(B_ * D_ * L_, 256), 256, 0, stream>>>(
      xz, (long)L_ * 2 * D_, 1, 2 * D_, conv_w, conv_b, xconv, 1);

  // 3) pe1 (1x1, leaky) ; pe2 (1x1, sigmoid) -> phase (B,4,L)
  gemm(xconv, 1, L_, (long)D_ * L_, pe1_w, D_, 0, 1,
       pemid, 1, L_, (long)48 * L_, pe1_b, 0, 1, FNULL, 0,
       L_, 48, D_, 1, B_);
  gemm(pemid, 1, L_, (long)48 * L_, pe2_w, 48, 0, 1,
       phase, 1, L_, (long)4 * L_, pe2_b, 0, 1, FNULL, 0,
       L_, 4, 48, 2, B_);

  // 4) illum mean; ig1 (3x3, leaky); ig2 (1x1) + softmax -> gate (B,4,L)
  illum_mean_kernel<<<cdiv(B_ * L_, 256), 256, 0, stream>>>(illum, I1);
  conv3x3_c1_kernel<<<cdiv(B_ * 48 * L_, 256), 256, 0, stream>>>(I1, ig1_w, ig1_b, igmid, 48, 1);
  gemm(igmid, 1, L_, (long)48 * L_, ig2_w, 48, 0, 1,
       gate, 1, L_, (long)4 * L_, ig2_b, 0, 1, FNULL, 0,
       L_, 4, 48, 0, B_);
  softmax_ch_kernel<<<cdiv(B_ * L_, 256), 256, 0, stream>>>(gate, 4);
  mul_kernel<<<cdiv(B_ * 4 * L_, 256), 256, 0, stream>>>(phase, gate, smb, B_ * 4 * L_);

  // 5) cross-scan gather
  cross_scan_kernel<<<cdiv(B_ * K_ * D_ * L_, 256), 256, 0, stream>>>(
      xconv, smb, scan_ids, xs, structS);

  // 6) x_proj per (b,k): (L,192) x (38,192)^T, row-scaled by struct -> xdbl (B,K,38,L)
  gemm(xs, 1, L_, (long)D_ * L_, xprojw, D_, (long)CP_ * D_, K_,
       xdbl, 1, L_, (long)CP_ * L_, FNULL, 0, 1, structS, L_,
       L_, CP_, D_, 0, B_ * K_);

  // 7) dt_proj per (b,k): dts (L,6) x (192,6)^T + bias, softplus -> delta (B,K,D,L)
  gemm(xdbl, 1, L_, (long)CP_ * L_, dtw, R_, (long)D_ * R_, K_,
       delta, 1, L_, (long)D_ * L_, dtb, D_, K_, FNULL, 0,
       L_, D_, R_, 3, B_ * K_);

  // 8) selective scan (+ Ds*xs)
  selective_scan_kernel<<<B_ * K_, D_, 0, stream>>>(xs, delta, xdbl, A_logs, Dsp, ysb);

  // 9) inverse gather + K-sum -> ysum (B,L,D); LN + silu(z) gate -> yact
  inv_merge_kernel<<<cdiv(B_ * L_ * D_, 256), 256, 0, stream>>>(ysb, inv_ids, ysum);
  ln_gate_kernel<<<B_ * L_, 256, 0, stream>>>(ysum, on_g, on_b, xz, yact);

  // 10) out_proj: (B*L,192) x (96,192)^T -> yout (B*L,96)
  gemm(yact, D_, 1, 0, outw, D_, 0, 1,
       yout, C_, 1, 0, FNULL, 0, 1, FNULL, 0,
       B_ * L_, C_, D_, 0, 1);

  // 11) phi_in (1x1 + bias): (L,96) x (192,96)^T -> Ff (B,192,L)
  gemm(yout, C_, 1, (long)L_ * C_, phi_in_w, C_, 0, 1,
       Ffb, 1, L_, (long)D_ * L_, phi_in_b, 0, 1, FNULL, 0,
       L_, D_, C_, 0, B_);

  // 12) router: 3x3 (1->8, gelu); 1x1 (8->3); softmax over E
  conv3x3_c1_kernel<<<cdiv(B_ * 8 * L_, 256), 256, 0, stream>>>(I1, r1w, r1b, rmid, 8, 4);
  gemm(rmid, 1, L_, (long)8 * L_, r2w, 8, 0, 1,
       alphab, 1, L_, (long)E_ * L_, r2b, 0, 1, FNULL, 0,
       L_, E_, 8, 0, B_);
  softmax_ch_kernel<<<cdiv(B_ * L_, 256), 256, 0, stream>>>(alphab, E_);

  // 13) fused experts -> Fout ; moeD depthwise + gelu -> moemid
  moe_expert_kernel<<<cdiv(B_ * D_ * L_, 256), 256, 0, stream>>>(Ffb, alphab, ew, eb, Foutb);
  dwconv3x3_kernel<<<cdiv(B_ * D_ * L_, 256), 256, 0, stream>>>(
      Foutb, (long)D_ * L_, L_, 1, moeDw, moeDb, moemid, 2);

  // 14) phi_out (1x1 + bias): (L,192) x (96,192)^T -> out (B,H,W,C)
  gemm(moemid, 1, L_, (long)D_ * L_, phow, D_, 0, 1,
       out, C_, 1, (long)L_ * C_, phob, 0, 1, FNULL, 0,
       L_, C_, D_, 0, B_);
}
